// DropBlock1DClassifier_48430051230172
// MI455X (gfx1250) — matrix operand order repaired and checked by reading, not compiled, not run on hardware
//
#include <hip/hip_runtime.h>

// ---- CDNA5 WMMA vector types ----
typedef __attribute__((ext_vector_type(16))) __bf16 v16bf;
typedef __attribute__((ext_vector_type(8)))  float  v8f;
typedef __attribute__((ext_vector_type(4)))  float  f32x4;

#define C_DIM     4096
#define NC        64
#define BLK_LEN   819
#define KCHUNKS   (C_DIM / 32)                 // 128 K-chunks of 32
#define NTILES    (NC / 16)                    // 4 N-tiles of 16
#define FRAGS     (KCHUNKS * NTILES)           // 512 B-fragments
#define PLANE_ELEMS ((size_t)FRAGS * 32)       // v16bf records per plane
#define PLANE_BYTES (PLANE_ELEMS * 32)         // 512 KB per plane (hi / lo)

// ---- helpers -------------------------------------------------------------

// Load 16 x-elements in the 16-bit A 16x32 per-lane layout matching B
// (lane<16: K {0..15}; lane>=16: K {16..31}), apply the
// drop-block mask, split into hi/lo bf16.  x is a 256 MB read-once stream
// (> 192 MB L2), so the loads are non-temporal to avoid evicting the
// L2-resident packed-W working set.
__device__ __forceinline__
void load_mask_split_A(const float* __restrict__ px, int oa, int s0,
                       v16bf& ahi, v16bf& alo)
{
    f32x4 aa[4];
    aa[0] = __builtin_nontemporal_load((const f32x4*)(px + oa));
    aa[1] = __builtin_nontemporal_load((const f32x4*)(px + oa + 4));
    aa[2] = __builtin_nontemporal_load((const f32x4*)(px + oa + 8));
    aa[3] = __builtin_nontemporal_load((const f32x4*)(px + oa + 12));
    #pragma unroll
    for (int q = 0; q < 4; ++q) {
        #pragma unroll
        for (int j = 0; j < 4; ++j) {
            const int idx = q * 4 + j;
            const int pos = oa + q * 4 + j;
            float v = aa[q][j];
            // blocked iff (unsigned)(pos - s0) < BLK_LEN : single compare
            v = ((unsigned)(pos - s0) < (unsigned)BLK_LEN) ? 0.0f : v;
            const __bf16 h = (__bf16)v;
            ahi[idx] = h;
            alo[idx] = (__bf16)(v - (float)h);
        }
    }
}

// Split 16 consecutive W floats into hi/lo bf16 (contiguous-k B 32x16 per-lane layout, matching A).
__device__ __forceinline__
void split_B16(const float* __restrict__ pw, v16bf& bhi, v16bf& blo)
{
    f32x4 bb[4];
    bb[0] = *(const f32x4*)(pw);
    bb[1] = *(const f32x4*)(pw + 4);
    bb[2] = *(const f32x4*)(pw + 8);
    bb[3] = *(const f32x4*)(pw + 12);
    #pragma unroll
    for (int q = 0; q < 4; ++q) {
        #pragma unroll
        for (int j = 0; j < 4; ++j) {
            const int idx = q * 4 + j;
            const float v = bb[q][j];
            const __bf16 h = (__bf16)v;
            bhi[idx] = h;
            blo[idx] = (__bf16)(v - (float)h);
        }
    }
}

#define WMMA_BF16(A, B, Cc) \
    __builtin_amdgcn_wmma_f32_16x16x32_bf16(false, (A), false, (B), (short)0, (Cc), false, false)

// ---- kernel 1: pack W into hi/lo bf16 B-fragments (done once, tiny) ------
// Fragment f = kc*NTILES + t ; record = 32 lanes x 32B, already in the exact
// per-lane VGPR layout the main kernel feeds as B (K = 16*half + s, as A).
__global__ __launch_bounds__(256)
void pack_W_hilo(const float* __restrict__ W,
                 v16bf* __restrict__ wh, v16bf* __restrict__ wl)
{
    const int lane = threadIdx.x & 31;
    const int wave = threadIdx.x >> 5;
    const int f    = blockIdx.x * 8 + wave;     // one wave per fragment
    const int kc   = f >> 2;
    const int t    = f & 3;
    const int mrow = lane & 15;
    const int half = lane >> 4;

    const float* pw = W + (size_t)(t * 16 + mrow) * C_DIM + kc * 32 + half * 16;
    v16bf hi, lo;
    split_B16(pw, hi, lo);
    wh[(size_t)f * 32 + lane] = hi;
    wl[(size_t)f * 32 + lane] = lo;
}

// ---- kernel 2: main GEMM, prepacked W, K-split x2 + LDS reduction --------
// Block = 256 threads = 8 waves = 4 row-tiles x 2 K-halves (2 waves/SIMD).
__global__ __launch_bounds__(256)
void dropblock_gemm_wmma_pre(const float* __restrict__ x,
                             const v16bf* __restrict__ wh,
                             const v16bf* __restrict__ wl,
                             const float* __restrict__ bias,
                             const int*   __restrict__ starts,
                             float*       __restrict__ out)
{
    __shared__ float red[4 * 32 * 32];          // 16 KB partial-sum buffer

    const int lane    = threadIdx.x & 31;
    const int wave    = threadIdx.x >> 5;
    const int rowtile = wave >> 1;              // 0..3
    const int khalf   = wave & 1;               // 0..1
    const int row0    = (blockIdx.x * 4 + rowtile) * 16;

    const int mrow = lane & 15;
    const int half = lane >> 4;

    const int    r  = row0 + mrow;
    const int    s0 = starts[r];
    const float* px = x + (size_t)r * C_DIM;

    v8f acc[NTILES];
    #pragma unroll
    for (int t = 0; t < NTILES; ++t) acc[t] = (v8f){};

    const int kbeg = khalf * (C_DIM / 2);
    const int kend = kbeg + (C_DIM / 2);

    for (int k0 = kbeg; k0 < kend; k0 += 32) {
        const int oa = k0 + half * 16;
        __builtin_prefetch(px + oa + 128, 0, 1);       // global_prefetch_b8

        v16bf ahi, alo;
        load_mask_split_A(px, oa, s0, ahi, alo);

        const int kc = k0 >> 5;
        #pragma unroll
        for (int t = 0; t < NTILES; ++t) {
            const size_t fo = (size_t)(kc * NTILES + t) * 32 + lane;
            const v16bf bhi = wh[fo];                  // 2x global_load_b128, L2-hit
            const v16bf blo = wl[fo];
            acc[t] = WMMA_BF16(ahi, bhi, acc[t]);
            acc[t] = WMMA_BF16(ahi, blo, acc[t]);
            acc[t] = WMMA_BF16(alo, bhi, acc[t]);
        }
    }

    // ---- cross-wave K reduction through LDS ----
    if (khalf) {
        #pragma unroll
        for (int t = 0; t < NTILES; ++t)
            #pragma unroll
            for (int i = 0; i < 8; ++i)
                red[((rowtile * 32) + (t * 8 + i)) * 32 + lane] = acc[t][i];
    }
    __syncthreads();
    if (!khalf) {
        #pragma unroll
        for (int t = 0; t < NTILES; ++t) {
            const int   col = t * 16 + mrow;
            const float bv  = bias[col];
            #pragma unroll
            for (int i = 0; i < 8; ++i) {
                const int row = row0 + half * 8 + i;   // C/D layout: M = i + 8*half
                const float p = red[((rowtile * 32) + (t * 8 + i)) * 32 + lane];
                __builtin_nontemporal_store(acc[t][i] + p + bv,
                                            out + (size_t)row * NC + col);
            }
        }
    }
}

// ---- kernel 3: fallback (ws too small) — on-the-fly W split --------------
__global__ __launch_bounds__(256)
void dropblock_gemm_wmma_fb(const float* __restrict__ x,
                            const float* __restrict__ W,
                            const float* __restrict__ bias,
                            const int*   __restrict__ starts,
                            float*       __restrict__ out)
{
    const int lane = threadIdx.x & 31;
    const int wave = threadIdx.x >> 5;
    const int row0 = (blockIdx.x * 8 + wave) * 16;
    const int mrow = lane & 15;
    const int half = lane >> 4;

    const int    r  = row0 + mrow;
    const int    s0 = starts[r];
    const float* px = x + (size_t)r * C_DIM;

    v8f acc[NTILES];
    #pragma unroll
    for (int t = 0; t < NTILES; ++t) acc[t] = (v8f){};

    for (int k0 = 0; k0 < C_DIM; k0 += 32) {
        const int oa = k0 + half * 16;
        __builtin_prefetch(px + oa + 128, 0, 1);

        v16bf ahi, alo;
        load_mask_split_A(px, oa, s0, ahi, alo);

        #pragma unroll
        for (int t = 0; t < NTILES; ++t) {
            const float* pw = W + (size_t)(t * 16 + mrow) * C_DIM + k0 + half * 16;
            v16bf bhi, blo;
            split_B16(pw, bhi, blo);
            acc[t] = WMMA_BF16(ahi, bhi, acc[t]);
            acc[t] = WMMA_BF16(ahi, blo, acc[t]);
            acc[t] = WMMA_BF16(alo, bhi, acc[t]);
        }
    }

    #pragma unroll
    for (int t = 0; t < NTILES; ++t) {
        const int   col = t * 16 + mrow;
        const float bv  = bias[col];
        #pragma unroll
        for (int i = 0; i < 8; ++i) {
            const int row = row0 + half * 8 + i;
            __builtin_nontemporal_store(acc[t][i] + bv,
                                        out + (size_t)row * NC + col);
        }
    }
}

// ---- launch --------------------------------------------------------------
extern "C" void kernel_launch(void* const* d_in, const int* in_sizes, int n_in,
                              void* d_out, int out_size, void* d_ws, size_t ws_size,
                              hipStream_t stream) {
    const float* x      = (const float*)d_in[0];
    const float* W      = (const float*)d_in[1];
    const float* b      = (const float*)d_in[2];
    const int*   starts = (const int*)  d_in[3];
    float*       out    = (float*)d_out;

    const int Brows = in_sizes[3];                     // 16384

    const size_t needed = 2 * PLANE_BYTES;             // 1 MB
    if (ws_size >= needed) {
        v16bf* wh = (v16bf*)d_ws;
        v16bf* wl = (v16bf*)((char*)d_ws + PLANE_BYTES);
        hipLaunchKernelGGL(pack_W_hilo, dim3(FRAGS / 8), dim3(256), 0, stream,
                           W, wh, wl);
        hipLaunchKernelGGL(dropblock_gemm_wmma_pre, dim3(Brows / 64), dim3(256),
                           0, stream, x, wh, wl, b, starts, out);
    } else {
        hipLaunchKernelGGL(dropblock_gemm_wmma_fb, dim3(Brows / 128), dim3(256),
                           0, stream, x, W, b, starts, out);
    }
}
